// AttentionDecoder_32152125177868
// MI455X (gfx1250) — compile-verified, matrix operand order checked by reading, not run on hardware
//
#include <hip/hip_runtime.h>

#define B_    64
#define T_    2048
#define DIN   256
#define H_    256
#define DOUT  128
#define OUTLEN 64

typedef __attribute__((ext_vector_type(16))) __bf16        v16bf;
typedef __attribute__((ext_vector_type(8)))  float         v8f;
typedef __attribute__((ext_vector_type(4)))  float         v4f;
typedef __attribute__((ext_vector_type(4)))  unsigned int  v4u;
typedef __attribute__((ext_vector_type(8)))  unsigned int  v8u;

__device__ __forceinline__ float rcp_fast(float x) {
    return __builtin_amdgcn_rcpf(x);           // single v_rcp_f32
}
__device__ __forceinline__ float tanh_fast(float x) {
    const float e = __expf(2.0f * x);          // hardware v_exp_f32
    return (e - 1.0f) * rcp_fast(e + 1.0f);
}
__device__ __forceinline__ float sigmoid_fast(float x) {
    return rcp_fast(1.0f + __expf(-x));
}
__device__ __forceinline__ float hsum4(v4f v) {
    return (v.x + v.y) + (v.z + v.w);
}

__device__ __forceinline__ v16bf load_frag16(const __bf16* p) {
    // two 16-byte loads of 8 contiguous bf16 each -> one v16bf fragment
    const v4u lo = *(const v4u*)(p);
    const v4u hi = *(const v4u*)(p + 16);
    const v8u pk = { lo.x, lo.y, lo.z, lo.w, hi.x, hi.y, hi.z, hi.w };
    return __builtin_bit_cast(v16bf, pk);
}

// ---------------------------------------------------------------------------
// K0: one-shot W_h_a (f32, row-major [n][k]) -> bf16 copy in workspace.
// ---------------------------------------------------------------------------
__global__ void __launch_bounds__(256)
k_cvt_w(const float* __restrict__ W, __bf16* __restrict__ Wb, int n)
{
    const int i = blockIdx.x * 256 + threadIdx.x;
    if (i < n) Wb[i] = (__bf16)W[i];           // native v_cvt bf16
}

// ---------------------------------------------------------------------------
// K1: twh[b,t] = sum_n wa1[n] * tanh( sum_k h[b,t,k] * W_h_a[n,k] )
// One WG = 64-row M-tile of the (B*T) x 256 GEMM.  h rows staged ONCE into
// LDS as bf16 (h read exactly 1x from HBM).  Each wave: 4 M-tiles x 4 N-tiles
// of accumulators so every B fragment (bf16 Wb copy, L2-resident) is reused
// by 4 WMMAs.  bf16 WMMA 16x16x32, K = 256 = 8 k-steps.
// ---------------------------------------------------------------------------
__global__ void __launch_bounds__(128)
k_twh(const float* __restrict__ h, const __bf16* __restrict__ Wb,
      const float* __restrict__ W_a, float* __restrict__ twh)
{
    const int tid   = threadIdx.x;
    const int wave  = tid >> 5;
    const int lane  = tid & 31;
    const int lrow  = lane & 15;        // row within M-tile / col within N-tile
    const int khalf = lane >> 4;        // lanes 16-31 hold the "other" K half
    const long rowBase = (long)blockIdx.x * 64;

    __shared__ __align__(16) __bf16 Als[64 * DIN];   // 32 KB: 64 h-rows, bf16
    __shared__ float twh_acc[64];

    // ---- cooperative stage: 64 rows = 16384 contiguous floats, f32x4 + pk-cvt
    {
        const float* src = h + rowBase * DIN;
        for (int i = tid; i < (64 * DIN) / 4; i += 128) {
            const v4f f = *(const v4f*)(src + (long)i * 4);
            Als[i * 4 + 0] = (__bf16)f.x;
            Als[i * 4 + 1] = (__bf16)f.y;
            Als[i * 4 + 2] = (__bf16)f.z;
            Als[i * 4 + 3] = (__bf16)f.w;
        }
        if (tid < 64) twh_acc[tid] = 0.0f;
    }
    __syncthreads();

    // ---- 4 M-tiles x 4 N-tiles of f32 accumulators per wave (128 VGPRs)
    v8f acc[4][4] = {};                 // acc[ntile][mtile]

    for (int kb = 0; kb < 8; ++kb) {
        const int k0 = kb * 32 + khalf * 8;
        // A fragments for the 4 M-tiles (ds_load_b128 pairs, no conversion)
        // lane l<16 : row l,    K = k0+{0..7} | k0+{16..23}
        // lane l>=16: row l-16, K = k0+{8..15}| k0+{24..31}  (khalf folds this)
        v16bf af[4];
        #pragma unroll
        for (int mt = 0; mt < 4; ++mt)
            af[mt] = load_frag16(Als + (mt * 16 + lrow) * DIN + k0);

        #pragma unroll
        for (int c = 0; c < 4; ++c) {
            const int n = (wave * 4 + c) * 16 + lrow;     // output column
            const v16bf bf = load_frag16(Wb + (long)n * DIN + k0); // B[k][n]=W[n][k]
            #pragma unroll
            for (int mt = 0; mt < 4; ++mt)
                acc[c][mt] = __builtin_amdgcn_wmma_f32_16x16x32_bf16(
                                 false, af[mt], false, bf,
                                 (short)0, acc[c][mt], false, false);
        }
    }

    // ---- epilogue: twh[m] += sum_n tanh(acc) * wa1[n]
    // C layout: VGPR v -> (M = v + 8*khalf, N = lrow)
    #pragma unroll
    for (int c = 0; c < 4; ++c) {
        const int n = (wave * 4 + c) * 16 + lrow;
        const float w1 = W_a[n];                          // wa1[n] = W_a[0, n]
        #pragma unroll
        for (int mt = 0; mt < 4; ++mt) {
            #pragma unroll
            for (int v = 0; v < 8; ++v) {
                const int m = mt * 16 + v + khalf * 8;
                atomicAdd(&twh_acc[m], tanh_fast(acc[c][mt][v]) * w1); // ds_add_f32
            }
        }
    }
    __syncthreads();
    if (tid < 64) twh[rowBase + tid] = twh_acc[tid];
}

// ---------------------------------------------------------------------------
// K2: a = softmax(twh[b,:]);  ci[b,d] = sum_t a[t] * h[b,t,d]
// (sa[b] is constant over t, so it cancels in the softmax — step-invariant.)
// ---------------------------------------------------------------------------
__global__ void __launch_bounds__(256)
k_attn_ctx(const float* __restrict__ h, const float* __restrict__ twh,
           float* __restrict__ ci)
{
    const int b   = blockIdx.x;
    const int tid = threadIdx.x;
    __shared__ float p[T_];
    __shared__ float red[256];

    const float* tw = twh + (long)b * T_;
    float mx = -1e30f;
    for (int t = tid; t < T_; t += 256) mx = fmaxf(mx, tw[t]);
    red[tid] = mx; __syncthreads();
    for (int s = 128; s > 0; s >>= 1) {
        if (tid < s) red[tid] = fmaxf(red[tid], red[tid + s]);
        __syncthreads();
    }
    mx = red[0]; __syncthreads();

    float sum = 0.0f;
    for (int t = tid; t < T_; t += 256) {
        float e = __expf(tw[t] - mx);
        p[t] = e; sum += e;
    }
    red[tid] = sum; __syncthreads();
    for (int s = 128; s > 0; s >>= 1) {
        if (tid < s) red[tid] += red[tid + s];
        __syncthreads();
    }
    const float inv = rcp_fast(red[0]);
    __syncthreads();

    // ci[b][tid] — coalesced column reads of h (L2-resident: h = 134 MB < 192 MB)
    const float* hb = h + (long)b * T_ * DIN;
    float acc = 0.0f;
    for (int t = 0; t < T_; ++t) {
        if (t + 16 < T_) __builtin_prefetch(&hb[(long)(t + 16) * DIN + tid], 0, 0);
        acc += p[t] * hb[(long)t * DIN + tid];
    }
    ci[b * DIN + tid] = acc * inv;
}

// ---------------------------------------------------------------------------
// K3: per-batch sequential decode: s0 = tanh(W_init h[b,0] + b_init), then
// 64 steps of GRU (x = [ci, y]) + RNN (x = [ci, s]) + softmax -> out[b,step,:]
// All dot products use 16-byte vector loads (rows are 16B-aligned).
// ---------------------------------------------------------------------------
__global__ void __launch_bounds__(256)
k_decode(const float* __restrict__ h,      const float* __restrict__ ci_g,
         const float* __restrict__ W_init, const float* __restrict__ b_init,
         const float* __restrict__ W_ih_g, const float* __restrict__ W_hh_g,
         const float* __restrict__ b_ih_g, const float* __restrict__ b_hh_g,
         const float* __restrict__ W_ih_r, const float* __restrict__ W_hh_r,
         const float* __restrict__ b_ih_r, const float* __restrict__ b_hh_r,
         float* __restrict__ out)
{
    const int b   = blockIdx.x;
    const int tid = threadIdx.x;
    __shared__ __align__(16) float ci[DIN];
    __shared__ __align__(16) float s[H_], s2[H_];
    __shared__ __align__(16) float y[DOUT];
    __shared__ __align__(16) float h0[DIN];
    __shared__ float red[DOUT], red2[DOUT];

    ci[tid] = ci_g[b * DIN + tid];
    h0[tid] = h[(long)b * T_ * DIN + tid];
    if (tid < DOUT) y[tid] = 0.0f;
    __syncthreads();

    // s0
    {
        const float* wr = W_init + (long)tid * DIN;
        v4f a4 = {0.f, 0.f, 0.f, 0.f};
        for (int d = 0; d < DIN; d += 4)
            a4 += *(const v4f*)(wr + d) * *(const v4f*)(h0 + d);
        s[tid] = tanh_fast(b_init[tid] + hsum4(a4));
    }
    __syncthreads();

    for (int step = 0; step < OUTLEN; ++step) {
        // ---- GRU: thread tid owns gate rows tid (r), tid+256 (z), tid+512 (n)
        const float* wr_r = W_ih_g + (long)tid * (DIN + DOUT);
        const float* wr_z = wr_r + (long)H_ * (DIN + DOUT);
        const float* wr_n = wr_z + (long)H_ * (DIN + DOUT);
        v4f ar = {0.f,0.f,0.f,0.f}, az = ar, an = ar;
        for (int d = 0; d < DIN; d += 4) {
            const v4f x = *(const v4f*)(ci + d);
            ar += *(const v4f*)(wr_r + d) * x;
            az += *(const v4f*)(wr_z + d) * x;
            an += *(const v4f*)(wr_n + d) * x;
        }
        for (int d = 0; d < DOUT; d += 4) {
            const v4f x = *(const v4f*)(y + d);
            ar += *(const v4f*)(wr_r + DIN + d) * x;
            az += *(const v4f*)(wr_z + DIN + d) * x;
            an += *(const v4f*)(wr_n + DIN + d) * x;
        }
        const float* hr_r = W_hh_g + (long)tid * H_;
        const float* hr_z = hr_r + (long)H_ * H_;
        const float* hr_n = hr_z + (long)H_ * H_;
        v4f br = {0.f,0.f,0.f,0.f}, bz = br, bn = br;
        for (int d = 0; d < H_; d += 4) {
            const v4f x = *(const v4f*)(s + d);
            br += *(const v4f*)(hr_r + d) * x;
            bz += *(const v4f*)(hr_z + d) * x;
            bn += *(const v4f*)(hr_n + d) * x;
        }
        const float gi_r = b_ih_g[tid]          + hsum4(ar);
        const float gi_z = b_ih_g[tid + H_]     + hsum4(az);
        const float gi_n = b_ih_g[tid + 2 * H_] + hsum4(an);
        const float gh_r = b_hh_g[tid]          + hsum4(br);
        const float gh_z = b_hh_g[tid + H_]     + hsum4(bz);
        const float gh_n = b_hh_g[tid + 2 * H_] + hsum4(bn);
        const float r = sigmoid_fast(gi_r + gh_r);
        const float z = sigmoid_fast(gi_z + gh_z);
        const float n = tanh_fast(gi_n + r * gh_n);
        s2[tid] = (1.0f - z) * n + z * s[tid];
        __syncthreads();

        // ---- RNN pre-activation (threads 0..127)
        float e = 0.0f;
        if (tid < DOUT) {
            const float* wi = W_ih_r + (long)tid * (DIN + H_);
            const float* wh = W_hh_r + (long)tid * DOUT;
            v4f p4 = {0.f,0.f,0.f,0.f};
            for (int d = 0; d < DIN; d += 4)
                p4 += *(const v4f*)(wi + d) * *(const v4f*)(ci + d);
            for (int d = 0; d < H_; d += 4)
                p4 += *(const v4f*)(wi + DIN + d) * *(const v4f*)(s2 + d);
            for (int d = 0; d < DOUT; d += 4)
                p4 += *(const v4f*)(wh + d) * *(const v4f*)(y + d);
            red[tid] = tanh_fast(b_ih_r[tid] + b_hh_r[tid] + hsum4(p4));
        }
        __syncthreads();

        // ---- softmax over 128
        if (tid < DOUT) red2[tid] = red[tid];
        __syncthreads();
        for (int w = 64; w > 0; w >>= 1) {
            if (tid < w) red2[tid] = fmaxf(red2[tid], red2[tid + w]);
            __syncthreads();
        }
        const float mx = red2[0]; __syncthreads();
        if (tid < DOUT) { e = __expf(red[tid] - mx); red2[tid] = e; }
        __syncthreads();
        for (int w = 64; w > 0; w >>= 1) {
            if (tid < w) red2[tid] += red2[tid + w];
            __syncthreads();
        }
        const float invZ = rcp_fast(red2[0]); __syncthreads();

        if (tid < DOUT) {
            const float yn = e * invZ;
            out[((long)b * OUTLEN + step) * DOUT + tid] = yn;
            y[tid] = yn;
        }
        s[tid] = s2[tid];
        __syncthreads();
    }
}

extern "C" void kernel_launch(void* const* d_in, const int* in_sizes, int n_in,
                              void* d_out, int out_size, void* d_ws, size_t ws_size,
                              hipStream_t stream)
{
    const float* h      = (const float*)d_in[0];
    const float* W_h_a  = (const float*)d_in[1];
    // d_in[2] (W_s_a), d_in[3] (b_s_a), W_a[256:] (wa2): cancel out —
    // softmax(twh + sa[:,None]) == softmax(twh) since sa is constant over t.
    const float* W_a    = (const float*)d_in[4];
    const float* W_init = (const float*)d_in[5];
    const float* b_init = (const float*)d_in[6];
    const float* W_ih_g = (const float*)d_in[7];
    const float* W_hh_g = (const float*)d_in[8];
    const float* b_ih_g = (const float*)d_in[9];
    const float* b_hh_g = (const float*)d_in[10];
    const float* W_ih_r = (const float*)d_in[11];
    const float* W_hh_r = (const float*)d_in[12];
    const float* b_ih_r = (const float*)d_in[13];
    const float* b_hh_r = (const float*)d_in[14];
    float* out = (float*)d_out;

    float*  twh = (float*)d_ws;                         // B*T floats   (512 KB)
    float*  ci  = twh + (size_t)B_ * T_;                // B*DIN floats (64 KB)
    __bf16* Wb  = (__bf16*)(ci + (size_t)B_ * DIN);     // H*DIN bf16   (128 KB)

    k_cvt_w<<<(H_ * DIN) / 256, 256, 0, stream>>>(W_h_a, Wb, H_ * DIN);
    k_twh<<<(B_ * T_) / 64, 128, 0, stream>>>(h, Wb, W_a, twh);
    k_attn_ctx<<<B_, 256, 0, stream>>>(h, twh, ci);
    k_decode<<<B_, 256, 0, stream>>>(h, ci, W_init, b_init,
                                     W_ih_g, W_hh_g, b_ih_g, b_hh_g,
                                     W_ih_r, W_hh_r, b_ih_r, b_hh_r, out);
}
